// PointConv_planar_11708080849163
// MI455X (gfx1250) — compile-verified, matrix operand order checked by reading, not run on hardware
//
#include <hip/hip_runtime.h>
#include <math.h>

// ---------------------------------------------------------------------------
// MI455X (gfx1250) PointConv forward.
// Compute is tiny (<5 GFLOP, <5MB footprint) -> latency bound. Strategy:
//  * density Gram matrix via V_WMMA_F32_16X16X4_F32 (exact f32)
//  * 1024->64 lin GEMM via V_WMMA_F32_16X16X32_F16 (f32 accumulate),
//    B operand streamed by the Tensor Data Mover into LDS (double buffered,
//    TENSORcnt-synchronized, TDM pad feature used to avoid LDS bank conflicts)
//  * everything else fused, LDS-resident, minimal kernel count.
// ---------------------------------------------------------------------------

typedef __attribute__((ext_vector_type(2)))  float        v2f;
typedef __attribute__((ext_vector_type(8)))  float        v8f;
typedef __attribute__((ext_vector_type(16))) _Float16     v16h;
typedef __attribute__((ext_vector_type(4)))  unsigned int u32x4;
typedef __attribute__((ext_vector_type(8)))  unsigned int u32x8;

#define BB   8
#define NN   4096
#define NP1  128
#define NS1  8
#define C1   64
#define AGGC 1024
#define BW1  0.1f
#define BW2  0.2f

struct P { const float* a[65]; };

enum {
  XYZ = 0,
  S1_BN_G, S1_BN_T, S1_BNL_G, S1_BNL_T, S1_CONV_B, S1_CONV_W,
  S1_D0_BG, S1_D0_BT, S1_D0_LB, S1_D0_LW,
  S1_D1_BG, S1_D1_BT, S1_D1_LB, S1_D1_LW,
  S1_D2_BG, S1_D2_BT, S1_D2_LB, S1_D2_LW,
  S1_LIN_B, S1_LIN_W,
  S1_W0_BG, S1_W0_BT, S1_W0_LB, S1_W0_LW,
  S1_W1_BG, S1_W1_BT, S1_W1_LB, S1_W1_LW,
  S1_W2_BG, S1_W2_BT, S1_W2_LB, S1_W2_LW,
  S2_BN_G, S2_BN_T, S2_BNL_G, S2_BNL_T, S2_CONV_B, S2_CONV_W,
  S2_D0_BG, S2_D0_BT, S2_D0_LB, S2_D0_LW,
  S2_D1_BG, S2_D1_BT, S2_D1_LB, S2_D1_LW,
  S2_D2_BG, S2_D2_BT, S2_D2_LB, S2_D2_LW,
  S2_LIN_B, S2_LIN_W,
  S2_W0_BG, S2_W0_BT, S2_W0_LB, S2_W0_LW,
  S2_W1_BG, S2_W1_BT, S2_W1_LB, S2_W1_LW,
  S2_W2_BG, S2_W2_BT, S2_W2_LB, S2_W2_LW,
  NPARAMS
};

__device__ __forceinline__ float reluf(float x)    { return x > 0.f ? x : 0.f; }
__device__ __forceinline__ float sigmoidf(float x) { return 1.f / (1.f + __expf(-x)); }

// ---------------------------------------------------------------------------
// Kernel 1: inverse density over all 4096 points, BW1.
// One wave (32 threads) per 16-row tile. d2 = |a|^2 + |b|^2 - 2*A.B, with the
// Gram part computed by V_WMMA_F32_16X16X4_F32 (K=3 padded to 4).
// ---------------------------------------------------------------------------
__global__ __launch_bounds__(32) void k_density1(const float* __restrict__ xyz,
                                                 float* __restrict__ invden) {
  const int wid  = blockIdx.x;          // b * 256 + rowtile
  const int b    = wid >> 8;
  const int rt   = wid & 255;
  const int lane = threadIdx.x;
  const int ln   = lane & 15;
  const int hh   = lane >> 4;

  const float* X = xyz + (size_t)b * 3 * NN;
  const int mrow = rt * 16 + ln;
  const float ax = X[0 * NN + mrow], ay = X[1 * NN + mrow], az = X[2 * NN + mrow];
  const float rowNorm = ax * ax + ay * ay + az * az;

  v2f afrag;
  if (hh == 0) { afrag[0] = ax; afrag[1] = ay; }
  else         { afrag[0] = az; afrag[1] = 0.f; }

  const float inv2bw2 = 1.0f / (2.0f * BW1 * BW1);
  float acc[8];
#pragma unroll
  for (int r = 0; r < 8; ++r) acc[r] = 0.f;

  for (int mt = 0; mt < NN / 16; ++mt) {
    const int ncol = mt * 16 + ln;
    const float bx = X[0 * NN + ncol], by = X[1 * NN + ncol], bz = X[2 * NN + ncol];
    const float colNorm = bx * bx + by * by + bz * bz;
    v2f bfrag;
    if (hh == 0) { bfrag[0] = bx; bfrag[1] = by; }
    else         { bfrag[0] = bz; bfrag[1] = 0.f; }

    v8f c = {};
    c = __builtin_amdgcn_wmma_f32_16x16x4_f32(false, afrag, false, bfrag,
                                              (short)0, c, false, false);
#pragma unroll
    for (int r = 0; r < 8; ++r) {
      const int M  = r + hh * 8;
      const float na = __shfl(rowNorm, M, 32);   // rows live on lanes 0..15 (dup 16..31)
      const float d2 = na + colNorm - 2.0f * c[r];
      acc[r] += __expf(-d2 * inv2bw2);
    }
  }

#pragma unroll
  for (int r = 0; r < 8; ++r) {
    float v = acc[r];
#pragma unroll
    for (int m = 1; m < 16; m <<= 1) v += __shfl_xor(v, m, 32);  // reduce within 16-lane half
    if (ln == 0) {
      const int row  = rt * 16 + r + hh * 8;
      const float dens = (v / (float)NN) / (2.5f * BW1);
      invden[(size_t)b * NN + row] = 1.0f / dens;
    }
  }
}

// ---------------------------------------------------------------------------
// Kernel 2: farthest point sampling (sequential 128 rounds), one block/batch.
// ---------------------------------------------------------------------------
__global__ __launch_bounds__(256) void k_fps(const float* __restrict__ xyz,
                                             int* __restrict__ fpsIdx,
                                             float* __restrict__ newXyz) {
  __shared__ float dist[NN];
  __shared__ float rv[256];
  __shared__ int   ri[256];
  __shared__ int   fpsL[NP1];
  __shared__ int   curFar;

  const int b = blockIdx.x;
  const int t = threadIdx.x;
  const float* X = xyz + (size_t)b * 3 * NN;

  for (int n = t; n < NN; n += 256) dist[n] = 1e10f;
  if (t == 0) curFar = 0;
  __syncthreads();

  for (int it = 0; it < NP1; ++it) {
    const int far = curFar;
    if (t == 0) { fpsIdx[b * NP1 + it] = far; fpsL[it] = far; }
    const float cx = X[far], cy = X[NN + far], cz = X[2 * NN + far];

    float bestV = -1.f; int bestI = 0;
    for (int n = t; n < NN; n += 256) {
      const float dx = X[n] - cx, dy = X[NN + n] - cy, dz = X[2 * NN + n] - cz;
      const float d  = dx * dx + dy * dy + dz * dz;
      const float dn = fminf(dist[n], d);
      dist[n] = dn;
      if (dn > bestV) { bestV = dn; bestI = n; }
    }
    rv[t] = bestV; ri[t] = bestI;
    __syncthreads();
    for (int s = 128; s > 0; s >>= 1) {
      if (t < s) {
        if (rv[t + s] > rv[t] || (rv[t + s] == rv[t] && ri[t + s] < ri[t])) {
          rv[t] = rv[t + s]; ri[t] = ri[t + s];
        }
      }
      __syncthreads();
    }
    if (t == 0) curFar = ri[0];
    __syncthreads();
  }

  if (t < NP1) {
    const int i = fpsL[t];
    newXyz[((size_t)b * NP1 + t) * 3 + 0] = X[i];
    newXyz[((size_t)b * NP1 + t) * 3 + 1] = X[NN + i];
    newXyz[((size_t)b * NP1 + t) * 3 + 2] = X[2 * NN + i];
  }
}

// ---------------------------------------------------------------------------
// Kernel 3: 8-NN of each FPS point; one thread per query, xyz cached in LDS.
// ---------------------------------------------------------------------------
__global__ __launch_bounds__(128) void k_knn(const float* __restrict__ xyz,
                                             const float* __restrict__ newXyz,
                                             int* __restrict__ knnIdx) {
  __shared__ float px[NN], py[NN], pz[NN];
  const int b = blockIdx.x;
  const int s = threadIdx.x;
  const float* X = xyz + (size_t)b * 3 * NN;
  for (int n = s; n < NN; n += 128) { px[n] = X[n]; py[n] = X[NN + n]; pz[n] = X[2 * NN + n]; }
  __syncthreads();

  const float qx = newXyz[((size_t)b * NP1 + s) * 3 + 0];
  const float qy = newXyz[((size_t)b * NP1 + s) * 3 + 1];
  const float qz = newXyz[((size_t)b * NP1 + s) * 3 + 2];

  float bd[NS1]; int bi[NS1];
#pragma unroll
  for (int k = 0; k < NS1; ++k) { bd[k] = 3.0e38f; bi[k] = 0; }

  for (int n = 0; n < NN; ++n) {
    const float dx = px[n] - qx, dy = py[n] - qy, dz = pz[n] - qz;
    const float d  = dx * dx + dy * dy + dz * dz;
    if (d < bd[NS1 - 1]) {
      int j = NS1 - 1;
#pragma unroll
      for (int u = NS1 - 1; u > 0; --u) {
        if (j > 0 && bd[j - 1] > d) { bd[j] = bd[j - 1]; bi[j] = bi[j - 1]; --j; }
      }
      bd[j] = d; bi[j] = n;
    }
  }
#pragma unroll
  for (int k = 0; k < NS1; ++k) knnIdx[((size_t)b * NP1 + s) * NS1 + k] = bi[k];
}

// ---------------------------------------------------------------------------
// Kernel 4: gather + conv(6->64) + density-net + weight-net + k-contraction.
// One 64-thread block per (b,s) group; everything in LDS.
// ---------------------------------------------------------------------------
__global__ __launch_bounds__(64) void k_sa1_group(P prm,
                                                  const float* __restrict__ newXyz,
                                                  const int* __restrict__ knnIdx,
                                                  const float* __restrict__ invden,
                                                  float* __restrict__ agg) {
  const int gid = blockIdx.x;         // b*128 + s
  const int b = gid >> 7;
  const int t = threadIdx.x;

  __shared__ float gx[NS1][3], pts[NS1][6], gden[NS1], dsv[NS1];
  __shared__ float wm[NS1][16], feat[NS1][C1];
  __shared__ float gmax;

  const float* X = prm.a[XYZ] + (size_t)b * 3 * NN;

  if (t < NS1) {
    const int k   = t;
    const int idx = knnIdx[(size_t)gid * NS1 + k];
    const float pxv = X[idx], pyv = X[NN + idx], pzv = X[2 * NN + idx];
    const float nx = newXyz[(size_t)gid * 3 + 0];
    const float ny = newXyz[(size_t)gid * 3 + 1];
    const float nz = newXyz[(size_t)gid * 3 + 2];
    gx[k][0] = pxv - nx; gx[k][1] = pyv - ny; gx[k][2] = pzv - nz;
    pts[k][0] = gx[k][0]; pts[k][1] = gx[k][1]; pts[k][2] = gx[k][2];
    pts[k][3] = pxv; pts[k][4] = pyv; pts[k][5] = pzv;
    gden[k] = invden[(size_t)b * NN + idx];
  }
  __syncthreads();
  if (t == 0) {
    float m = gden[0];
    for (int k = 1; k < NS1; ++k) m = fmaxf(m, gden[k]);
    gmax = m;
  }
  __syncthreads();

  if (t < NS1) {
    const int k = t;
    // density net: 1 -> 16 -> 8 -> 1 (relu, relu, sigmoid)
    const float x0 = gden[k] / gmax;
    float h1[16], h2[8];
    for (int j = 0; j < 16; ++j)
      h1[j] = reluf((prm.a[S1_D0_LW][j] * x0 + prm.a[S1_D0_LB][j]) * prm.a[S1_D0_BG][j] + prm.a[S1_D0_BT][j]);
    for (int j = 0; j < 8; ++j) {
      float acc = prm.a[S1_D1_LB][j];
      for (int i = 0; i < 16; ++i) acc += prm.a[S1_D1_LW][j * 16 + i] * h1[i];
      h2[j] = reluf(acc * prm.a[S1_D1_BG][j] + prm.a[S1_D1_BT][j]);
    }
    {
      float acc = prm.a[S1_D2_LB][0];
      for (int i = 0; i < 8; ++i) acc += prm.a[S1_D2_LW][i] * h2[i];
      dsv[k] = sigmoidf(acc * prm.a[S1_D2_BG][0] + prm.a[S1_D2_BT][0]);
    }
    // weight net: 3 -> 8 -> 8 -> 16 (relu x3)
    float w1[8], w2[8];
    for (int j = 0; j < 8; ++j) {
      float acc = prm.a[S1_W0_LB][j];
      for (int i = 0; i < 3; ++i) acc += prm.a[S1_W0_LW][j * 3 + i] * gx[k][i];
      w1[j] = reluf(acc * prm.a[S1_W0_BG][j] + prm.a[S1_W0_BT][j]);
    }
    for (int j = 0; j < 8; ++j) {
      float acc = prm.a[S1_W1_LB][j];
      for (int i = 0; i < 8; ++i) acc += prm.a[S1_W1_LW][j * 8 + i] * w1[i];
      w2[j] = reluf(acc * prm.a[S1_W1_BG][j] + prm.a[S1_W1_BT][j]);
    }
    for (int j = 0; j < 16; ++j) {
      float acc = prm.a[S1_W2_LB][j];
      for (int i = 0; i < 8; ++i) acc += prm.a[S1_W2_LW][j * 8 + i] * w2[i];
      wm[k][j] = reluf(acc * prm.a[S1_W2_BG][j] + prm.a[S1_W2_BT][j]);
    }
  }
  __syncthreads();

  {  // conv 6 -> 64, BN, relu, * ds   (thread t = output channel)
    const int c = t;
    float wrow[6];
    for (int i = 0; i < 6; ++i) wrow[i] = prm.a[S1_CONV_W][c * 6 + i];
    const float bb = prm.a[S1_CONV_B][c], g = prm.a[S1_BN_G][c], tt = prm.a[S1_BN_T][c];
    for (int k = 0; k < NS1; ++k) {
      float acc = bb;
      for (int i = 0; i < 6; ++i) acc += wrow[i] * pts[k][i];
      feat[k][c] = reluf(acc * g + tt) * dsv[k];
    }
  }
  __syncthreads();

  {  // agg[c][j] = sum_k feat[k][c] * w[k][j]  (c = t, 16 j's each)
    const int c = t;
    float* out = agg + (size_t)gid * AGGC + c * 16;
    for (int j = 0; j < 16; ++j) {
      float acc = 0.f;
      for (int k = 0; k < NS1; ++k) acc += feat[k][c] * wm[k][j];
      out[j] = acc;
    }
  }
}

// ---------------------------------------------------------------------------
// Tensor Data Mover: issue a 2D tile load (16 rows x 32 floats, row stride
// 1024 floats) from global into LDS, with LDS padding of 4 DWORDs every
// 32 DWORDs (-> LDS row stride 36 floats, bank-conflict free for the 64-bank
// LDS when 16 lanes read 16 different rows).
// D# layout per CDNA5 ISA 8.3/8.4 (group0 128b, group1 256b), 2D form with
// VADDR2/VADDR3 = NULL.
// ---------------------------------------------------------------------------
#define TDM_TILE_X   32          // floats per row per tile
#define TDM_TILE_Y   16          // rows
#define TDM_LDS_PITCH 36         // floats (32 + 4 pad DWORDs)

__device__ __forceinline__ void tdm_load_b_tile(unsigned ldsOff, const float* gptr) {
  const unsigned long long ga = (unsigned long long)(uintptr_t)gptr;
  u32x4 g0;
  g0[0] = 1u;                                                   // count=1, user desc
  g0[1] = ldsOff;                                               // lds_addr (bytes)
  g0[2] = (unsigned)ga;                                         // global_addr[31:0]
  g0[3] = (unsigned)((ga >> 32) & 0x01FFFFFFull) | (2u << 30);  // addr[56:32] | type=2
  u32x8 g1;
  g1[0] = (2u << 16)        // data_size = 2 (4 bytes)
        | (1u << 20)        // pad_enable
        | (4u << 22)        // pad_interval code 4 = every 32 DWORDs
        | (3u << 25);       // pad_amount  code 3 = 4 DWORDs
  g1[1] = (AGGC & 0xFFFFu) << 16;                    // tensor_dim0[15:0]
  g1[2] = (AGGC >> 16) | ((64u & 0xFFFFu) << 16);    // tensor_dim0[31:16] | tensor_dim1[15:0]
  g1[3] = (unsigned)(TDM_TILE_X) << 16;              // tensor_dim1[31:16]=0 | tile_dim0
  g1[4] = (unsigned)TDM_TILE_Y;                      // tile_dim1 | tile_dim2=0
  g1[5] = (unsigned)AGGC;                            // tensor_dim0_stride[31:0]
  g1[6] = 0u;                                        // stride0[47:32] | stride1[15:0]
  g1[7] = 0u;                                        // stride1[47:16]
  asm volatile("tensor_load_to_lds %0, %1" :: "s"(g0), "s"(g1) : "memory");
}

// ---------------------------------------------------------------------------
// Kernel 5: lin (128x1024)x(1024x64) per batch via V_WMMA_F32_16X16X32_F16.
// One wave per 16x16 output tile. B operand (16 rows of lin.w) streamed by
// the TDM into double-buffered LDS tiles (TENSORcnt pipeline); A operand via
// float4 global loads + prefetch. Fused bias/BN/ReLU epilogue.
// ---------------------------------------------------------------------------
__global__ __launch_bounds__(32) void k_sa1_lin(const float* __restrict__ agg,
                                                const float* __restrict__ W,
                                                const float* __restrict__ bias,
                                                const float* __restrict__ g,
                                                const float* __restrict__ tt,
                                                float* __restrict__ l1pts) {
  __shared__ __align__(16) float Btile[2][TDM_TILE_Y * TDM_LDS_PITCH];

  const int wid = blockIdx.x;            // b*32 + mt*4 + nt
  const int b  = wid >> 5;
  const int mt = (wid >> 2) & 7;
  const int nt = wid & 3;
  const int lane = threadIdx.x;
  const int ln = lane & 15;
  const int hh = lane >> 4;

  const float* Arow  = agg + ((size_t)b * NP1 + mt * 16 + ln) * AGGC;
  const float* Bbase = W + (size_t)(nt * 16) * AGGC;     // 16 consecutive rows of lin.w
  const float4* A4 = (const float4*)Arow;

  const unsigned ldsOff0 = (unsigned)(uintptr_t)&Btile[0][0];
  const unsigned ldsOff1 = (unsigned)(uintptr_t)&Btile[1][0];

  v8f acc = {};

  // software pipeline: tile kt covers K = [kt*32, kt*32+32)
  tdm_load_b_tile(ldsOff0, Bbase + 0 * TDM_TILE_X);

  for (int kt = 0; kt < AGGC / TDM_TILE_X - 1; ++kt) {
    // issue next tile into the other buffer (max 2 outstanding)
    tdm_load_b_tile((kt & 1) ? ldsOff0 : ldsOff1, Bbase + (kt + 1) * TDM_TILE_X);
    // prefetch next A chunk
    __builtin_prefetch(Arow + (kt + 1) * TDM_TILE_X, 0, 1);
    // wait for current tile (in-order TENSORcnt: <=1 means tile kt landed)
    __builtin_amdgcn_s_wait_tensorcnt(1);

    const int k0 = kt * TDM_TILE_X + hh * 8;
    const int i0 = k0 >> 2;
    const float4 a0 = A4[i0], a1 = A4[i0 + 1], a2 = A4[i0 + 4], a3 = A4[i0 + 5];
    const float* Brow = &Btile[kt & 1][ln * TDM_LDS_PITCH + hh * 8];
    const float4 b0 = *(const float4*)(Brow);
    const float4 b1 = *(const float4*)(Brow + 4);
    const float4 b2 = *(const float4*)(Brow + 16);
    const float4 b3 = *(const float4*)(Brow + 20);
    v16h av, bv;
    av[0]  = (_Float16)a0.x; av[1]  = (_Float16)a0.y; av[2]  = (_Float16)a0.z; av[3]  = (_Float16)a0.w;
    av[4]  = (_Float16)a1.x; av[5]  = (_Float16)a1.y; av[6]  = (_Float16)a1.z; av[7]  = (_Float16)a1.w;
    av[8]  = (_Float16)a2.x; av[9]  = (_Float16)a2.y; av[10] = (_Float16)a2.z; av[11] = (_Float16)a2.w;
    av[12] = (_Float16)a3.x; av[13] = (_Float16)a3.y; av[14] = (_Float16)a3.z; av[15] = (_Float16)a3.w;
    bv[0]  = (_Float16)b0.x; bv[1]  = (_Float16)b0.y; bv[2]  = (_Float16)b0.z; bv[3]  = (_Float16)b0.w;
    bv[4]  = (_Float16)b1.x; bv[5]  = (_Float16)b1.y; bv[6]  = (_Float16)b1.z; bv[7]  = (_Float16)b1.w;
    bv[8]  = (_Float16)b2.x; bv[9]  = (_Float16)b2.y; bv[10] = (_Float16)b2.z; bv[11] = (_Float16)b2.w;
    bv[12] = (_Float16)b3.x; bv[13] = (_Float16)b3.y; bv[14] = (_Float16)b3.z; bv[15] = (_Float16)b3.w;
    acc = __builtin_amdgcn_wmma_f32_16x16x32_f16(false, av, false, bv,
                                                 (short)0, acc, false, false);
  }

  {  // peeled last tile (kt = 31), wait for all tensor traffic
    const int kt = AGGC / TDM_TILE_X - 1;
    __builtin_amdgcn_s_wait_tensorcnt(0);
    const int k0 = kt * TDM_TILE_X + hh * 8;
    const int i0 = k0 >> 2;
    const float4 a0 = A4[i0], a1 = A4[i0 + 1], a2 = A4[i0 + 4], a3 = A4[i0 + 5];
    const float* Brow = &Btile[kt & 1][ln * TDM_LDS_PITCH + hh * 8];
    const float4 b0 = *(const float4*)(Brow);
    const float4 b1 = *(const float4*)(Brow + 4);
    const float4 b2 = *(const float4*)(Brow + 16);
    const float4 b3 = *(const float4*)(Brow + 20);
    v16h av, bv;
    av[0]  = (_Float16)a0.x; av[1]  = (_Float16)a0.y; av[2]  = (_Float16)a0.z; av[3]  = (_Float16)a0.w;
    av[4]  = (_Float16)a1.x; av[5]  = (_Float16)a1.y; av[6]  = (_Float16)a1.z; av[7]  = (_Float16)a1.w;
    av[8]  = (_Float16)a2.x; av[9]  = (_Float16)a2.y; av[10] = (_Float16)a2.z; av[11] = (_Float16)a2.w;
    av[12] = (_Float16)a3.x; av[13] = (_Float16)a3.y; av[14] = (_Float16)a3.z; av[15] = (_Float16)a3.w;
    bv[0]  = (_Float16)b0.x; bv[1]  = (_Float16)b0.y; bv[2]  = (_Float16)b0.z; bv[3]  = (_Float16)b0.w;
    bv[4]  = (_Float16)b1.x; bv[5]  = (_Float16)b1.y; bv[6]  = (_Float16)b1.z; bv[7]  = (_Float16)b1.w;
    bv[8]  = (_Float16)b2.x; bv[9]  = (_Float16)b2.y; bv[10] = (_Float16)b2.z; bv[11] = (_Float16)b2.w;
    bv[12] = (_Float16)b3.x; bv[13] = (_Float16)b3.y; bv[14] = (_Float16)b3.z; bv[15] = (_Float16)b3.w;
    acc = __builtin_amdgcn_wmma_f32_16x16x32_f16(false, av, false, bv,
                                                 (short)0, acc, false, false);
  }

  const int o = nt * 16 + ln;
  const float bb = bias[o], gg = g[o], t0 = tt[o];
#pragma unroll
  for (int r = 0; r < 8; ++r) {
    const int srow = mt * 16 + r + hh * 8;
    l1pts[((size_t)b * NP1 + srow) * C1 + o] = reluf((acc[r] + bb) * gg + t0);
  }
}

// ---------------------------------------------------------------------------
// Kernel 6: entire SA2 (group_all over 128 points) in one block per batch.
// ---------------------------------------------------------------------------
__global__ __launch_bounds__(128) void k_sa2(P prm,
                                             const float* __restrict__ newXyz,
                                             const float* __restrict__ l1pts,
                                             float* __restrict__ out) {
  const int b = blockIdx.x;
  const int t = threadIdx.x;   // point index n = 0..127

  __shared__ float pxyz[NP1][3];
  __shared__ float red[128];
  __shared__ float featL[NP1][16];
  __shared__ float wL[NP1][16];
  __shared__ float aggL[256];

  const float qx = newXyz[((size_t)b * NP1 + t) * 3 + 0];
  const float qy = newXyz[((size_t)b * NP1 + t) * 3 + 1];
  const float qz = newXyz[((size_t)b * NP1 + t) * 3 + 2];
  pxyz[t][0] = qx; pxyz[t][1] = qy; pxyz[t][2] = qz;
  __syncthreads();

  // density over the 128 l1 points, BW2
  const float inv2bw2 = 1.f / (2.f * BW2 * BW2);
  float ssum = 0.f;
  for (int m = 0; m < NP1; ++m) {
    const float dx = pxyz[m][0] - qx, dy = pxyz[m][1] - qy, dz = pxyz[m][2] - qz;
    ssum += __expf(-(dx * dx + dy * dy + dz * dz) * inv2bw2);
  }
  const float dens  = (ssum / (float)NP1) / (2.5f * BW2);
  const float myinv = 1.f / dens;

  red[t] = myinv;
  __syncthreads();
  for (int s = 64; s > 0; s >>= 1) {
    if (t < s) red[t] = fmaxf(red[t], red[t + s]);
    __syncthreads();
  }
  const float gmax = red[0];

  // density net 2
  float dsv;
  {
    const float x0 = myinv / gmax;
    float h1[16], h2[8];
    for (int j = 0; j < 16; ++j)
      h1[j] = reluf((prm.a[S2_D0_LW][j] * x0 + prm.a[S2_D0_LB][j]) * prm.a[S2_D0_BG][j] + prm.a[S2_D0_BT][j]);
    for (int j = 0; j < 8; ++j) {
      float acc = prm.a[S2_D1_LB][j];
      for (int i = 0; i < 16; ++i) acc += prm.a[S2_D1_LW][j * 16 + i] * h1[i];
      h2[j] = reluf(acc * prm.a[S2_D1_BG][j] + prm.a[S2_D1_BT][j]);
    }
    float acc = prm.a[S2_D2_LB][0];
    for (int i = 0; i < 8; ++i) acc += prm.a[S2_D2_LW][i] * h2[i];
    dsv = sigmoidf(acc * prm.a[S2_D2_BG][0] + prm.a[S2_D2_BT][0]);
  }

  // conv2: 67 -> 16, BN, relu, * ds.  Inputs: [qx,qy,qz, l1pts(64)]
  const float* f64 = l1pts + ((size_t)b * NP1 + t) * C1;
  for (int c = 0; c < 16; ++c) {
    const float* wr = prm.a[S2_CONV_W] + c * 67;
    float acc = prm.a[S2_CONV_B][c] + wr[0] * qx + wr[1] * qy + wr[2] * qz;
    for (int i = 0; i < 64; ++i) acc += wr[3 + i] * f64[i];
    featL[t][c] = reluf(acc * prm.a[S2_BN_G][c] + prm.a[S2_BN_T][c]) * dsv;
  }

  // wnet2 on absolute coords (group_all does not recenter)
  {
    float w1[8], w2[8];
    const float gxi[3] = { qx, qy, qz };
    for (int j = 0; j < 8; ++j) {
      float acc = prm.a[S2_W0_LB][j];
      for (int i = 0; i < 3; ++i) acc += prm.a[S2_W0_LW][j * 3 + i] * gxi[i];
      w1[j] = reluf(acc * prm.a[S2_W0_BG][j] + prm.a[S2_W0_BT][j]);
    }
    for (int j = 0; j < 8; ++j) {
      float acc = prm.a[S2_W1_LB][j];
      for (int i = 0; i < 8; ++i) acc += prm.a[S2_W1_LW][j * 8 + i] * w1[i];
      w2[j] = reluf(acc * prm.a[S2_W1_BG][j] + prm.a[S2_W1_BT][j]);
    }
    for (int j = 0; j < 16; ++j) {
      float acc = prm.a[S2_W2_LB][j];
      for (int i = 0; i < 8; ++i) acc += prm.a[S2_W2_LW][j * 8 + i] * w2[i];
      wL[t][j] = reluf(acc * prm.a[S2_W2_BG][j] + prm.a[S2_W2_BT][j]);
    }
  }
  __syncthreads();

  // agg[c*16+j] = sum_n featL[n][c] * wL[n][j]   (256 entries, 2 per thread)
  for (int e = t; e < 256; e += 128) {
    const int c = e >> 4, j = e & 15;
    float acc = 0.f;
    for (int n = 0; n < NP1; ++n) acc += featL[n][c] * wL[n][j];
    aggL[e] = acc;
  }
  __syncthreads();

  // lin2: 256 -> 16, BN, relu -> final output (B,16)
  if (t < 16) {
    const float* wr = prm.a[S2_LIN_W] + t * 256;
    float acc = prm.a[S2_LIN_B][t];
    for (int i = 0; i < 256; ++i) acc += wr[i] * aggL[i];
    out[b * 16 + t] = reluf(acc * prm.a[S2_BNL_G][t] + prm.a[S2_BNL_T][t]);
  }
}

// ---------------------------------------------------------------------------
extern "C" void kernel_launch(void* const* d_in, const int* in_sizes, int n_in,
                              void* d_out, int out_size, void* d_ws, size_t ws_size,
                              hipStream_t stream) {
  (void)in_sizes; (void)out_size; (void)ws_size;

  P prm;
  for (int i = 0; i < 65; ++i) prm.a[i] = (i < n_in) ? (const float*)d_in[i] : nullptr;

  float* ws     = (float*)d_ws;
  float* invden = ws;                      // 8*4096            = 32768 f
  float* newXyz = ws + 32768;              // 8*128*3           = 3072 f
  float* l1pts  = ws + 35840;              // 8*128*64          = 65536 f
  float* agg    = ws + 101376;             // 8*128*1024        = 1048576 f
  int*   fpsIdx = (int*)(ws + 1149952);    // 8*128             = 1024 i
  int*   knnIdx = fpsIdx + 1024;           // 8*128*8           = 8192 i

  k_density1 <<<BB * (NN / 16), 32, 0, stream>>>(prm.a[XYZ], invden);
  k_fps      <<<BB, 256, 0, stream>>>(prm.a[XYZ], fpsIdx, newXyz);
  k_knn      <<<BB, 128, 0, stream>>>(prm.a[XYZ], newXyz, knnIdx);
  k_sa1_group<<<BB * NP1, 64, 0, stream>>>(prm, newXyz, knnIdx, invden, agg);
  k_sa1_lin  <<<BB * 32, 32, 0, stream>>>(agg, prm.a[S1_LIN_W], prm.a[S1_LIN_B],
                                          prm.a[S1_BNL_G], prm.a[S1_BNL_T], l1pts);
  k_sa2      <<<BB, 128, 0, stream>>>(prm, newXyz, l1pts, (float*)d_out);
}
